// EntNet_40295383171411
// MI455X (gfx1250) — matrix operand order repaired and checked by reading, not compiled, not run on hardware
//
#include <hip/hip_runtime.h>

typedef __attribute__((ext_vector_type(16))) _Float16 v16h;
typedef __attribute__((ext_vector_type(8)))  _Float16 v8h;
typedef __attribute__((ext_vector_type(8)))  float    v8f;

#define DD   128   // embedding dim
#define PAD  64
#define MM   2048  // memory slots
#define NF   4096  // steps
#define WSL  64    // columns per workgroup slab
#define NWG  (MM / WSL)   // 32
#define HST  129   // padded f32 column stride
#define H16ST 136  // padded f16 column stride (272B rows -> 16B aligned, bank-spread)

// Branch-free fast transcendentals for the scan's dependent chain.
// tanh(x) = 1 - 2/(e^{2x}+1): e^{2x}->inf => 1, ->0 => -1 (no NaN, no branches)
__device__ __forceinline__ float fast_tanh(float x) {
    float t = __expf(2.f * x);
    return 1.f - 2.f * __builtin_amdgcn_rcpf(t + 1.f);
}
__device__ __forceinline__ float fast_sigmoid(float x) {
    float t = __expf(-x);
    return __builtin_amdgcn_rcpf(1.f + t);
}

// ---------------------------------------------------------------------------
// Kernel 1: S[n,d] = sum_p F[d,p] * E_s[n,d,p]   (one pass over 128MB E_s)
// ---------------------------------------------------------------------------
__global__ void s_reduce_kernel(const float* __restrict__ E,
                                const float* __restrict__ F,
                                float* __restrict__ S) {
    int n = blockIdx.x;
    int d = threadIdx.x;            // 128 threads
    const float* e = E + ((size_t)n * DD + d) * PAD;
    const float* f = F + (size_t)d * PAD;
    float acc = 0.f;
#pragma unroll 8
    for (int p = 0; p < PAD; ++p) acc += e[p] * f[p];
    S[n * DD + d] = acc;
}

// ---------------------------------------------------------------------------
// Kernel 2: ZS[n,:] = Z @ S[n,:]
// ---------------------------------------------------------------------------
__global__ void zs_kernel(const float* __restrict__ S,
                          const float* __restrict__ Z,
                          float* __restrict__ ZS) {
    __shared__ float sS[DD];
    int n = blockIdx.x, d = threadIdx.x;   // 128 threads
    sS[d] = S[n * DD + d];
    __syncthreads();
    const float* zr = Z + d * DD;
    float acc = 0.f;
#pragma unroll 8
    for (int k = 0; k < DD; ++k) acc += zr[k] * sS[k];
    ZS[n * DD + d] = acc;
}

// ---------------------------------------------------------------------------
// Kernel 3: persistent scan. 32 WGs x 256 threads (8 wave32).
// Each WG owns a 64-column slab of H in LDS; X lives in VGPRs as WMMA
// A-fragments; per step: gate, stage H->f16, WMMA X@H, gated update,
// column renorm. Only per-step HBM traffic: 1KB of S/ZS rows.
// ---------------------------------------------------------------------------
__global__ __launch_bounds__(256)
void scan_kernel(const float* __restrict__ S,  const float* __restrict__ ZS,
                 const float* __restrict__ Xg, const float* __restrict__ Yg,
                 const float* __restrict__ W0, const float* __restrict__ H0,
                 float* __restrict__ Hout) {
    __shared__ float    Hs [WSL][HST];    // H slab, column-major  (33KB)
    __shared__ float    W0s[WSL][HST];    // W0 slab               (33KB)
    __shared__ float    YWs[WSL][HST];    // (Y@W0) slab           (33KB)
    __shared__ _Float16 H16[WSL][H16ST];  // f16 staging for WMMA B (17KB)
    __shared__ float sS[DD], sZ[DD];
    __shared__ float gpart[WSL][4];
    __shared__ float gate[WSL];
    __shared__ float colSS[WSL];

    const int tid  = threadIdx.x;
    const int lane = tid & 31;
    const int wave = tid >> 5;            // 0..7 -> row tile of output
    const int colBase = blockIdx.x * WSL;

    // ---- load H0 / W0 slabs (H0,W0 are (D,M) row-major) ----
    for (int i = tid; i < WSL * DD; i += 256) {
        int c = i >> 7, r = i & 127;
        Hs [c][r] = H0[(size_t)r * MM + colBase + c];
        W0s[c][r] = W0[(size_t)r * MM + colBase + c];
    }
    __syncthreads();

    // ---- precompute YW slab: YWs[c][r] = sum_k Y[r,k] * W0[k, colBase+c] ----
    for (int i = tid; i < WSL * DD; i += 256) {
        int c = i >> 7, r = i & 127;
        const float* yr = Yg + r * DD;
        float acc = 0.f;
#pragma unroll 8
        for (int k = 0; k < DD; ++k) acc += yr[k] * W0s[c][k];
        YWs[c][r] = acc;
    }

    // ---- preload X as WMMA A-fragments (16-bit A 16x32 ISA layout) ----
    // lane<16: VGPR0-3 hold K=0..7, VGPR4-7 hold K=16..23 ; lane>=16: +8
    v16h afrag[4];
    {
        int mrow  = wave * 16 + (lane & 15);
        int khalf = (lane >> 4) * 8;
#pragma unroll
        for (int kt = 0; kt < 4; ++kt)
#pragma unroll
            for (int e = 0; e < 16; ++e) {
                int kk = kt * 32 + khalf + (e & 7) + ((e & 8) << 1);
                afrag[kt][e] = (_Float16)Xg[mrow * DD + kk];
            }
    }
    if (tid < WSL) colSS[tid] = 0.f;
    __syncthreads();

    const int nlo   = lane & 15;                      // output column within tile
    const int vrow0 = wave * 16 + ((lane >> 4) << 3); // first output row of lane

    // =================== sequential scan ===================
    for (int n = 0; n < NF; ++n) {
        // stage s and Z@s rows
        if (tid < DD)            sS[tid]      = S [n * DD + tid];
        else if (tid < 2 * DD)   sZ[tid - DD] = ZS[n * DD + (tid - DD)];
        __syncthreads();

        // gate partials: gpre[c] = sum_r s[r]*(H[c][r] + W0[c][r])
        {
            int c = tid >> 2, part = tid & 3, r0 = part * 32;
            float acc = 0.f;
#pragma unroll 8
            for (int r = r0; r < r0 + 32; ++r)
                acc += sS[r] * (Hs[c][r] + W0s[c][r]);
            gpart[c][part] = acc;
        }
        // stage old H -> f16 for WMMA B operands
        for (int i = tid; i < WSL * DD; i += 256) {
            int c = i >> 7, r = i & 127;
            H16[c][r] = (_Float16)Hs[c][r];
        }
        __syncthreads();
        if (tid < WSL) {
            float g = gpart[tid][0] + gpart[tid][1] + gpart[tid][2] + gpart[tid][3];
            gate[tid] = fast_sigmoid(g);
        }
        __syncthreads();

        // WMMA: acc = X(16x128) @ Hold(128x16) per tile, then gated update
#pragma unroll
        for (int ct = 0; ct < 4; ++ct) {
            v8f acc = {};
            int bcol = ct * 16 + nlo;
#pragma unroll
            for (int kt = 0; kt < 4; ++kt) {
                int kbase = kt * 32 + ((lane >> 4) << 3);   // B: lane half h holds K=8h..8h+7, 16+8h..23+8h
                const v8h* bp = reinterpret_cast<const v8h*>(&H16[bcol][kbase]);
                v8h blo = bp[0];
                v8h bhi = bp[2];
                v16h b = __builtin_shufflevector(blo, bhi,
                          0,1,2,3,4,5,6,7,8,9,10,11,12,13,14,15);
                acc = __builtin_amdgcn_wmma_f32_16x16x32_f16(
                          false, afrag[kt], false, b, (short)0, acc, false, false);
            }
            // C layout: VGPR v -> row vrow0+v, column bcol (lane-owned elements)
            float g = gate[bcol];
            float ss = 0.f;
#pragma unroll
            for (int v = 0; v < 8; ++v) {
                int r = vrow0 + v;
                float hn = fast_tanh(acc[v] + YWs[bcol][r] + sZ[r]);
                float hu = Hs[bcol][r] + g * hn;   // in-place: (r,c) owned by this lane
                Hs[bcol][r] = hu;
                ss += hu * hu;
            }
            atomicAdd(&colSS[bcol], ss);           // ds_add_f32 cross-wave reduce
        }
        __syncthreads();

        // per-column renormalize
        {
            int c = tid >> 2, part = tid & 3, r0 = part * 32;
            float scale = __builtin_amdgcn_rsqf(colSS[c]);
#pragma unroll 8
            for (int r = r0; r < r0 + 32; ++r) Hs[c][r] *= scale;
        }
        __syncthreads();
        if (tid < WSL) colSS[tid] = 0.f;   // made visible by next step's barrier
    }

    // ---- write final H slab ----
    __syncthreads();
    for (int i = tid; i < WSL * DD; i += 256) {
        int c = i >> 7, r = i & 127;
        Hout[(size_t)r * MM + colBase + c] = Hs[c][r];
    }
}

// ---------------------------------------------------------------------------
// Kernel 4: readout.  q=(F*Q).sum ; p=softmax(q^T H) ; u=H p ;
// ans = (R/||R_row||) @ sigmoid(q + K u)
// ---------------------------------------------------------------------------
__global__ void final_kernel(const float* __restrict__ F, const float* __restrict__ Q,
                             const float* __restrict__ H, const float* __restrict__ Kg,
                             const float* __restrict__ R, float* __restrict__ out) {
    __shared__ float q[DD];
    __shared__ float p[MM];
    __shared__ float red[256];
    __shared__ float u[DD];
    __shared__ float vv[DD];
    int tid = threadIdx.x;   // 256 threads

    if (tid < DD) {
        const float* f  = F + tid * PAD;
        const float* qq = Q + tid * PAD;
        float a = 0.f;
        for (int i = 0; i < PAD; ++i) a += f[i] * qq[i];
        q[tid] = a;
    }
    __syncthreads();

    float lmax = -1e30f;
    for (int m = tid; m < MM; m += 256) {
        float a = 0.f;
        for (int d = 0; d < DD; ++d) a += q[d] * H[(size_t)d * MM + m];
        p[m] = a;
        lmax = fmaxf(lmax, a);
    }
    red[tid] = lmax; __syncthreads();
    for (int s = 128; s > 0; s >>= 1) {
        if (tid < s) red[tid] = fmaxf(red[tid], red[tid + s]);
        __syncthreads();
    }
    lmax = red[0];
    float lsum = 0.f;
    for (int m = tid; m < MM; m += 256) {
        float e = __expf(p[m] - lmax);
        p[m] = e; lsum += e;
    }
    __syncthreads();
    red[tid] = lsum; __syncthreads();
    for (int s = 128; s > 0; s >>= 1) {
        if (tid < s) red[tid] += red[tid + s];
        __syncthreads();
    }
    float inv = 1.f / red[0];

    if (tid < DD) {
        const float* hr = H + (size_t)tid * MM;
        float a = 0.f;
        for (int m = 0; m < MM; ++m) a += p[m] * hr[m];
        u[tid] = a * inv;
    }
    __syncthreads();
    if (tid < DD) {
        const float* kr = Kg + tid * DD;
        float a = 0.f;
        for (int k = 0; k < DD; ++k) a += kr[k] * u[k];
        float x = q[tid] + a;
        vv[tid] = 1.f / (1.f + __expf(-x));
    }
    __syncthreads();
    if (tid < DD) {
        const float* rr = R + tid * DD;
        float nrm = 0.f, a = 0.f;
        for (int k = 0; k < DD; ++k) { nrm += rr[k] * rr[k]; a += rr[k] * vv[k]; }
        out[tid] = a * rsqrtf(nrm);
    }
}

// ---------------------------------------------------------------------------
extern "C" void kernel_launch(void* const* d_in, const int* in_sizes, int n_in,
                              void* d_out, int out_size, void* d_ws, size_t ws_size,
                              hipStream_t stream) {
    const float* E  = (const float*)d_in[0];   // (NF, D, PAD)
    const float* Q  = (const float*)d_in[1];   // (D, PAD)
    const float* F  = (const float*)d_in[2];   // (D, PAD)
    const float* X  = (const float*)d_in[3];   // (D, D)
    const float* Y  = (const float*)d_in[4];   // (D, D)
    const float* Z  = (const float*)d_in[5];   // (D, D)
    const float* R  = (const float*)d_in[6];   // (D, D)
    const float* K  = (const float*)d_in[7];   // (D, D)
    const float* H0 = (const float*)d_in[8];   // (D, M)
    const float* W0 = (const float*)d_in[9];   // (D, M)

    float* Sbuf  = (float*)d_ws;               // NF*D   (2 MB)
    float* ZSbuf = Sbuf  + (size_t)NF * DD;    // NF*D   (2 MB)
    float* Hfin  = ZSbuf + (size_t)NF * DD;    // D*M    (1 MB)

    s_reduce_kernel<<<NF, DD, 0, stream>>>(E, F, Sbuf);
    zs_kernel      <<<NF, DD, 0, stream>>>(Sbuf, Z, ZSbuf);
    scan_kernel    <<<NWG, 256, 0, stream>>>(Sbuf, ZSbuf, X, Y, W0, H0, Hfin);
    final_kernel   <<<1, 256, 0, stream>>>(F, Q, Hfin, K, R, (float*)d_out);
}
